// minGRU_25383256720183
// MI455X (gfx1250) — matrix operand order repaired and checked by reading, not compiled, not run on hardware
//
#include <hip/hip_runtime.h>
#include <hip/hip_bf16.h>
#include <math.h>
#include <stdint.h>

typedef __attribute__((ext_vector_type(16))) __bf16 v16bf;
typedef __attribute__((ext_vector_type(8)))  __bf16 bf16x8;
typedef __attribute__((ext_vector_type(8)))  float  v8f;
typedef __attribute__((ext_vector_type(4)))  unsigned int u32x4;
typedef __attribute__((ext_vector_type(4)))  int i32x4;
typedef __attribute__((ext_vector_type(8)))  int i32x8;

#define BATCH 4
#define SEQ   4096
#define KDIM  1024
#define DI    1024
#define MTOT  (BATCH * SEQ)   // 16384
#define CH    64              // scan chunks
#define CL    64              // chunk length (CH*CL == SEQ)

// ---------------------------------------------------------------- TDM support probe
#if defined(__has_builtin)
#if __has_builtin(__builtin_amdgcn_tensor_load_to_lds) && __has_builtin(__builtin_amdgcn_s_wait_tensorcnt)
#define HAVE_TDM 1
#endif
#endif
#ifndef HAVE_TDM
#define HAVE_TDM 0
#endif

#if HAVE_TDM
// Issue one TDM 2D tile load: rows x 32 bf16 elements, row stride KDIM elems,
// packed contiguously into LDS at lds_addr. D# built per CDNA5 ISA ch.8.
__device__ __forceinline__ void tdm_load_tile(const __bf16* gsrc, unsigned lds_addr,
                                              unsigned rows, unsigned tot_rows) {
    unsigned long long ga = (unsigned long long)(uintptr_t)gsrc;
    u32x4 g0;
    g0[0] = 1u;                                               // count=1, user mode
    g0[1] = lds_addr;                                         // LDS byte address
    g0[2] = (unsigned)ga;                                     // global_addr[31:0]
    g0[3] = (unsigned)((ga >> 32) & 0x1ffffffull) | 0x80000000u; // addr[56:32] | type=2
    i32x8 g1;
    g1[0] = 0x00010000;                  // data_size=1 (2 bytes), mask=0, no pad/iter
    g1[1] = (int)(KDIM << 16);           // tensor_dim0[15:0] = 1024 (bits 79:48)
    g1[2] = (int)((tot_rows & 0xffffu) << 16);                // dim0 hi=0 | tensor_dim1 lo
    g1[3] = (int)(((tot_rows >> 16) & 0xffffu) | (32u << 16)); // tensor_dim1 hi | tile_dim0=32
    g1[4] = (int)rows;                   // tile_dim1 = rows, tile_dim2 = 0
    g1[5] = KDIM;                        // tensor_dim0_stride[31:0] = 1024
    g1[6] = 0;                           // stride hi | tensor_dim1_stride lo
    g1[7] = 0;
    i32x4 z4 = {0, 0, 0, 0};
#if __clang_major__ >= 23
    i32x8 z8 = {0, 0, 0, 0, 0, 0, 0, 0};
    __builtin_amdgcn_tensor_load_to_lds(g0, g1, z4, z4, z8, 0);
#else
    __builtin_amdgcn_tensor_load_to_lds(g0, g1, z4, z4, 0);
#endif
}
#endif

// ---------------------------------------------------------------- helpers
__device__ __forceinline__ float softplus_f(float x) {
    return fmaxf(x, 0.f) + log1pf(__expf(-fabsf(x)));
}
__device__ __forceinline__ float logaddexp_f(float a, float b) {
    float mx = fmaxf(a, b);
    return mx + log1pf(__expf(-fabsf(a - b)));
}

// ---------------------------------------------------------------- fp32 -> bf16
__global__ void cvt_f32_bf16(const float* __restrict__ src,
                             __bf16* __restrict__ dst, int n) {
    int i = blockIdx.x * blockDim.x + threadIdx.x;
    int stride = gridDim.x * blockDim.x;
    for (; i < n; i += stride) dst[i] = (__bf16)src[i];
}

// ---------------------------------------------------------------- GEMM1 + minGRU epilogue
// Macro tile 128 rows x 64 channels; wave tile 32x32 with hidden/gate pairing.
// Double-buffered LDS tiles; TDM staging when available.
__global__ __launch_bounds__(256) void gemm1_mingru(
    const __bf16* __restrict__ xb, const __bf16* __restrict__ whb,
    float* __restrict__ lcb, float* __restrict__ lvb)
{
    __shared__ __align__(16) __bf16 As[2][128 * 32];
    __shared__ __align__(16) __bf16 Hs[2][64 * 32];
    __shared__ __align__(16) __bf16 Gs[2][64 * 32];

    const int tid  = threadIdx.x;
    const int lane = tid & 31;
    const int wave = tid >> 5;
    const int wm   = wave & 3;
    const int wn   = wave >> 2;
    const int m0   = blockIdx.x * 128;
    const int n0   = blockIdx.y * 64;

#if !HAVE_TDM
    const int arow = tid >> 1;
    const int acol = (tid & 1) * 16;
    const int wrow = tid >> 2;
    const int wcol = (tid & 3) * 8;
#endif

    v8f acch[2][2] = {};
    v8f accg[2][2] = {};

    const int mrow = lane & 15;
    const int aklo = (lane < 16) ? 0 : 8;
    const int bkb  = (lane < 16) ? 0 : 16;
    const int rm   = wm * 32;
    const int cn   = wn * 32;

    union Frag { v16bf v; bf16x8 h[2]; };
    const int NK = KDIM / 32;

    auto stage = [&](int i) {
        const int buf = i & 1;
        const int k0  = i * 32;
#if HAVE_TDM
        if (wave == 0)
            tdm_load_tile(&xb[(size_t)m0 * KDIM + k0],
                          (unsigned)(uintptr_t)&As[buf][0], 128, MTOT);
        else if (wave == 1)
            tdm_load_tile(&whb[(size_t)n0 * KDIM + k0],
                          (unsigned)(uintptr_t)&Hs[buf][0], 64, 2 * DI);
        else if (wave == 2)
            tdm_load_tile(&whb[(size_t)(DI + n0) * KDIM + k0],
                          (unsigned)(uintptr_t)&Gs[buf][0], 64, 2 * DI);
#else
        {
            const __bf16* s = &xb[(size_t)(m0 + arow) * KDIM + k0 + acol];
            __bf16* d = &As[buf][arow * 32 + acol];
            *reinterpret_cast<bf16x8*>(d)     = *reinterpret_cast<const bf16x8*>(s);
            *reinterpret_cast<bf16x8*>(d + 8) = *reinterpret_cast<const bf16x8*>(s + 8);
        }
        *reinterpret_cast<bf16x8*>(&Hs[buf][wrow * 32 + wcol]) =
            *reinterpret_cast<const bf16x8*>(&whb[(size_t)(n0 + wrow) * KDIM + k0 + wcol]);
        *reinterpret_cast<bf16x8*>(&Gs[buf][wrow * 32 + wcol]) =
            *reinterpret_cast<const bf16x8*>(&whb[(size_t)(DI + n0 + wrow) * KDIM + k0 + wcol]);
#endif
    };

    stage(0);
    for (int i = 0; i < NK; ++i) {
        if (i + 1 < NK) stage(i + 1);
#if HAVE_TDM
        if (wave < 3) {
            if (i + 1 < NK) __builtin_amdgcn_s_wait_tensorcnt(1);
            else            __builtin_amdgcn_s_wait_tensorcnt(0);
        }
#endif
        __syncthreads();
        const int buf = i & 1;

        Frag aa[2], bh[2], bg[2];
        #pragma unroll
        for (int a = 0; a < 2; ++a) {
            const int ab = (rm + a * 16 + mrow) * 32;
            aa[a].h[0] = *reinterpret_cast<const bf16x8*>(&As[buf][ab + bkb]);
            aa[a].h[1] = *reinterpret_cast<const bf16x8*>(&As[buf][ab + bkb + 8]);
        }
        #pragma unroll
        for (int t = 0; t < 2; ++t) {
            const int nb = (cn + t * 16 + mrow) * 32;
            bh[t].h[0] = *reinterpret_cast<const bf16x8*>(&Hs[buf][nb + bkb]);
            bh[t].h[1] = *reinterpret_cast<const bf16x8*>(&Hs[buf][nb + bkb + 8]);
            bg[t].h[0] = *reinterpret_cast<const bf16x8*>(&Gs[buf][nb + bkb]);
            bg[t].h[1] = *reinterpret_cast<const bf16x8*>(&Gs[buf][nb + bkb + 8]);
        }

        #pragma unroll
        for (int a = 0; a < 2; ++a)
            #pragma unroll
            for (int t = 0; t < 2; ++t) {
                acch[a][t] = __builtin_amdgcn_wmma_f32_16x16x32_bf16(
                    false, aa[a].v, false, bh[t].v, (short)0, acch[a][t], false, false);
                accg[a][t] = __builtin_amdgcn_wmma_f32_16x16x32_bf16(
                    false, aa[a].v, false, bg[t].v, (short)0, accg[a][t], false, false);
            }
        __syncthreads();
    }

    // epilogue: minGRU log-space precompute
    #pragma unroll
    for (int a = 0; a < 2; ++a) {
        const int mbase = m0 + rm + a * 16 + ((lane < 16) ? 0 : 8);
        #pragma unroll
        for (int t = 0; t < 2; ++t) {
            const int c = n0 + cn + t * 16 + mrow;
            #pragma unroll
            for (int r = 0; r < 8; ++r) {
                float h = acch[a][t][r], g = accg[a][t][r];
                float lg = (h >= 0.f) ? __logf(h + 0.5f) : -softplus_f(-h);
                size_t idx = (size_t)(mbase + r) * DI + c;
                lcb[idx] = -softplus_f(g);
                lvb[idx] = lg - softplus_f(-g);
            }
        }
    }
}

// ---------------------------------------------------------------- chunked scan
__global__ void scan_chunk_reduce(const float* __restrict__ lcb, const float* __restrict__ lvb,
                                  float* __restrict__ Ac, float* __restrict__ Vc)
{
    int idx = blockIdx.x * blockDim.x + threadIdx.x;
    int d     = idx & (DI - 1);
    int chunk = (idx >> 10) & (CH - 1);
    int b     = idx >> 16;
    size_t base = ((size_t)b * SEQ + (size_t)chunk * CL) * DI + d;
    float A = 0.f, V = -INFINITY;
    for (int s = 0; s < CL; ++s) {
        size_t i = base + (size_t)s * DI;
        float c = lcb[i], v = lvb[i];
        A += c;
        V = logaddexp_f(V + c, v);
    }
    Ac[idx] = A;
    Vc[idx] = V;
}

__global__ void scan_chunk_carry(const float* __restrict__ Ac, const float* __restrict__ Vc,
                                 float* __restrict__ Lin)
{
    int idx = blockIdx.x * blockDim.x + threadIdx.x;   // 0..4095
    int b = idx >> 10;
    int d = idx & (DI - 1);
    float lh = -INFINITY;
    for (int ch = 0; ch < CH; ++ch) {
        size_t j = ((size_t)b * CH + ch) * DI + d;
        Lin[j] = lh;
        lh = logaddexp_f(Ac[j] + lh, Vc[j]);
    }
}

__global__ void scan_chunk_apply(const float* __restrict__ lcb, const float* __restrict__ lvb,
                                 const float* __restrict__ Lin, __bf16* __restrict__ hb)
{
    int idx = blockIdx.x * blockDim.x + threadIdx.x;
    int d     = idx & (DI - 1);
    int chunk = (idx >> 10) & (CH - 1);
    int b     = idx >> 16;
    size_t base = ((size_t)b * SEQ + (size_t)chunk * CL) * DI + d;
    float lh = Lin[idx];
    for (int s = 0; s < CL; ++s) {
        size_t i = base + (size_t)s * DI;
        lh = logaddexp_f(lcb[i] + lh, lvb[i]);
        hb[i] = (__bf16)__expf(lh);
    }
}

// ---------------------------------------------------------------- GEMM2: out = h @ W_out^T
// Macro tile 128x128; wave tile 32x64; double-buffered LDS; TDM staging when available.
__global__ __launch_bounds__(256) void gemm2_out(
    const __bf16* __restrict__ hb, const __bf16* __restrict__ wob,
    float* __restrict__ out)
{
    __shared__ __align__(16) __bf16 As[2][128 * 32];
    __shared__ __align__(16) __bf16 Bs[2][128 * 32];

    const int tid  = threadIdx.x;
    const int lane = tid & 31;
    const int wave = tid >> 5;
    const int wm   = wave & 3;
    const int wn   = wave >> 2;
    const int m0   = blockIdx.x * 128;
    const int n0   = blockIdx.y * 128;

#if !HAVE_TDM
    const int srow = tid >> 1;
    const int scol = (tid & 1) * 16;
#endif

    v8f acc[2][4] = {};

    const int mrow = lane & 15;
    const int aklo = (lane < 16) ? 0 : 8;
    const int bkb  = (lane < 16) ? 0 : 16;
    const int rm   = wm * 32;
    const int cn   = wn * 64;

    union Frag { v16bf v; bf16x8 h[2]; };
    const int NK = DI / 32;

    auto stage = [&](int i) {
        const int buf = i & 1;
        const int k0  = i * 32;
#if HAVE_TDM
        if (wave == 0)
            tdm_load_tile(&hb[(size_t)m0 * DI + k0],
                          (unsigned)(uintptr_t)&As[buf][0], 128, MTOT);
        else if (wave == 1)
            tdm_load_tile(&wob[(size_t)n0 * DI + k0],
                          (unsigned)(uintptr_t)&Bs[buf][0], 128, DI);
#else
        {
            const __bf16* s = &hb[(size_t)(m0 + srow) * DI + k0 + scol];
            __bf16* d = &As[buf][srow * 32 + scol];
            *reinterpret_cast<bf16x8*>(d)     = *reinterpret_cast<const bf16x8*>(s);
            *reinterpret_cast<bf16x8*>(d + 8) = *reinterpret_cast<const bf16x8*>(s + 8);
        }
        {
            const __bf16* s = &wob[(size_t)(n0 + srow) * DI + k0 + scol];
            __bf16* d = &Bs[buf][srow * 32 + scol];
            *reinterpret_cast<bf16x8*>(d)     = *reinterpret_cast<const bf16x8*>(s);
            *reinterpret_cast<bf16x8*>(d + 8) = *reinterpret_cast<const bf16x8*>(s + 8);
        }
#endif
    };

    stage(0);
    for (int i = 0; i < NK; ++i) {
        if (i + 1 < NK) stage(i + 1);
#if HAVE_TDM
        if (wave < 2) {
            if (i + 1 < NK) __builtin_amdgcn_s_wait_tensorcnt(1);
            else            __builtin_amdgcn_s_wait_tensorcnt(0);
        }
#endif
        __syncthreads();
        const int buf = i & 1;

        Frag aa[2], bb[4];
        #pragma unroll
        for (int a = 0; a < 2; ++a) {
            const int ab = (rm + a * 16 + mrow) * 32;
            aa[a].h[0] = *reinterpret_cast<const bf16x8*>(&As[buf][ab + bkb]);
            aa[a].h[1] = *reinterpret_cast<const bf16x8*>(&As[buf][ab + bkb + 8]);
        }
        #pragma unroll
        for (int t = 0; t < 4; ++t) {
            const int nb = (cn + t * 16 + mrow) * 32;
            bb[t].h[0] = *reinterpret_cast<const bf16x8*>(&Bs[buf][nb + bkb]);
            bb[t].h[1] = *reinterpret_cast<const bf16x8*>(&Bs[buf][nb + bkb + 8]);
        }

        #pragma unroll
        for (int a = 0; a < 2; ++a)
            #pragma unroll
            for (int t = 0; t < 4; ++t)
                acc[a][t] = __builtin_amdgcn_wmma_f32_16x16x32_bf16(
                    false, aa[a].v, false, bb[t].v, (short)0, acc[a][t], false, false);
        __syncthreads();
    }

    #pragma unroll
    for (int a = 0; a < 2; ++a) {
        const int mbase = m0 + rm + a * 16 + ((lane < 16) ? 0 : 8);
        #pragma unroll
        for (int t = 0; t < 4; ++t) {
            const int c = n0 + cn + t * 16 + mrow;
            #pragma unroll
            for (int r = 0; r < 8; ++r)
                out[(size_t)(mbase + r) * DI + c] = acc[a][t][r];
        }
    }
}

// ---------------------------------------------------------------- launch
extern "C" void kernel_launch(void* const* d_in, const int* in_sizes, int n_in,
                              void* d_out, int out_size, void* d_ws, size_t ws_size,
                              hipStream_t stream)
{
    const float* x    = (const float*)d_in[0];   // [4,4096,1024]
    const float* Whg  = (const float*)d_in[1];   // [2048,1024]
    const float* Wout = (const float*)d_in[2];   // [1024,1024]
    float* out = (float*)d_out;                  // [4,4096,1024]

    char* ws = (char*)d_ws;
    size_t off = 0;
    auto take = [&](size_t bytes) -> void* {
        void* p = ws + off;
        off += (bytes + 255) & ~(size_t)255;
        return p;
    };
    __bf16* xb  = (__bf16*)take((size_t)MTOT * KDIM * 2);        // 32 MB
    __bf16* whb = (__bf16*)take((size_t)2 * DI * KDIM * 2);      //  4 MB
    __bf16* wob = (__bf16*)take((size_t)DI * DI * 2);            //  2 MB
    float*  lcb = (float*)take((size_t)MTOT * DI * 4);           // 64 MB
    float*  lvb = (float*)take((size_t)MTOT * DI * 4);           // 64 MB
    __bf16* hb  = (__bf16*)take((size_t)MTOT * DI * 2);          // 32 MB
    float*  Ac  = (float*)take((size_t)BATCH * CH * DI * 4);     //  1 MB
    float*  Vc  = (float*)take((size_t)BATCH * CH * DI * 4);     //  1 MB
    float*  Lin = (float*)take((size_t)BATCH * CH * DI * 4);     //  1 MB

    cvt_f32_bf16<<<4096, 256, 0, stream>>>(x,    xb,  MTOT * KDIM);
    cvt_f32_bf16<<<2048, 256, 0, stream>>>(Whg,  whb, 2 * DI * KDIM);
    cvt_f32_bf16<<<1024, 256, 0, stream>>>(Wout, wob, DI * DI);

    gemm1_mingru<<<dim3(MTOT / 128, DI / 64), 256, 0, stream>>>(xb, whb, lcb, lvb);

    scan_chunk_reduce<<<(BATCH * CH * DI) / 256, 256, 0, stream>>>(lcb, lvb, Ac, Vc);
    scan_chunk_carry <<<(BATCH * DI) / 256,       256, 0, stream>>>(Ac, Vc, Lin);
    scan_chunk_apply <<<(BATCH * CH * DI) / 256, 256, 0, stream>>>(lcb, lvb, Lin, hb);

    gemm2_out<<<dim3(MTOT / 128, DI / 128), 256, 0, stream>>>(hb, wob, out);
}
